// UnetMACReconNetEncDecKM_5420248727879
// MI455X (gfx1250) — compile-verified, matrix operand order checked by reading, not run on hardware
//
#include <hip/hip_runtime.h>
#include <hip/hip_bf16.h>
#include <math.h>

typedef __attribute__((ext_vector_type(16))) __bf16 v16bf;
typedef __attribute__((ext_vector_type(8)))  float  v8f;

static __device__ __forceinline__ __bf16 f2bf(float f) {
  union { float f; unsigned u; } v; v.f = f;
  unsigned r = (v.u + 0x7FFFu + ((v.u >> 16) & 1u)) >> 16;
  unsigned short h = (unsigned short)r;
  __bf16 o; __builtin_memcpy(&o, &h, 2); return o;
}
static __device__ __forceinline__ int imin(int a, int b) { return a < b ? a : b; }
static __device__ __forceinline__ int imax(int a, int b) { return a > b ? a : b; }

// ===========================================================================
// All activations are NHWC:  t[b][p][c], flat = (b*HW + p)*C + c.
// Conv inputs (Ci>=32) are NHWC zero-halo-padded bf16.
// ===========================================================================

// NHWC f32 -> NHWC padded bf16 (+1 halo, zero border)
__global__ __launch_bounds__(256) void pad_nhwc(const float* __restrict__ x,
    __bf16* __restrict__ xp, int C, int H, int W, int N)
{
  int i = blockIdx.x * 256 + threadIdx.x; if (i >= N) return;
  int Wp = W + 2, Hp = H + 2;
  int c  = i % C;  int t = i / C;
  int xo = t % Wp; t /= Wp;
  int yo = t % Hp; int b = t / Hp;
  float v = 0.f;
  if (xo >= 1 && xo <= W && yo >= 1 && yo <= H)
    v = x[(((size_t)b * H + (yo - 1)) * W + (xo - 1)) * C + c];
  xp[i] = f2bf(v);
}

// ---------------------------------------------------------------------------
// First layer (Ci=1, Co=32) direct conv: stage 288 modulated weights in LDS.
// grid = (HW/256, 1, B)
// ---------------------------------------------------------------------------
__global__ __launch_bounds__(256) void conv0_direct(
    const float* __restrict__ x, const float* __restrict__ Wt,
    const float* __restrict__ bias, const float* __restrict__ mod,
    float* __restrict__ y, int H, int W)
{
  __shared__ float mw[288];
  __shared__ float bs[32];
  int b = blockIdx.z, tid = threadIdx.x;
  if (tid < 288) mw[tid] = Wt[tid] * mod[(size_t)b * 288 + tid];
  if (tid < 32)  bs[tid] = bias[tid];
  __syncthreads();
  int HW = H * W;
  int p  = blockIdx.x * 256 + tid;
  int py = p / W, px = p % W;
  float taps[9];
#pragma unroll
  for (int r = 0; r < 9; ++r) {
    int dy = (r >= 3) + (r >= 6);
    int dx = r - dy * 3;
    int yy = py + dy - 1, xx = px + dx - 1;
    taps[r] = (yy >= 0 && yy < H && xx >= 0 && xx < W)
                ? x[(size_t)b * HW + yy * W + xx] : 0.f;
  }
  float* yb = y + ((size_t)b * HW + p) * 32;
  for (int co = 0; co < 32; ++co) {
    float acc = bs[co];
#pragma unroll
    for (int r = 0; r < 9; ++r) acc += mw[co * 9 + r] * taps[r];
    yb[co] = acc;
  }
}

// ---------------------------------------------------------------------------
// Modulated 3x3 conv, implicit GEMM on WMMA bf16 (fp32 accum), NHWC, Ci>=32.
// K reordered k = r*Ci + ci; each 32-chunk has uniform r -> uniform (dy,dx).
// Per-lane element offsets hoisted; per-chunk offset is wave-uniform SALU.
// Per chunk: 2 ds_load_b128 (A) + 8 global_load_b128 (B) + 4 WMMA.
// grid = (max(HW/512,1), Co/16, B), dynamic LDS = 16*K*2 bytes. K % 32 == 0.
// ---------------------------------------------------------------------------
extern __shared__ __bf16 Alds[];

__global__ __launch_bounds__(256) void modconv3_wmma(
    const __bf16* __restrict__ xp, const float* __restrict__ Wt,
    const float* __restrict__ bias, const float* __restrict__ mod,
    float* __restrict__ y, int Ci, int ciShift, int Co, int H, int wdShift, int K)
{
  const int Wd  = 1 << wdShift;
  const int HW  = H << wdShift;
  const int Wp  = Wd + 2;
  const int b      = blockIdx.z;
  const int coBase = blockIdx.y * 16;
  const int tid  = threadIdx.x;
  const int lane = tid & 31, wave = tid >> 5;
  const int ciMask = Ci - 1;

  // ---- stage modulated weights into LDS, reordered k = r*Ci + ci ----
  {
    const float* Wb = Wt  + (size_t)coBase * K;
    const float* Mb = mod + ((size_t)b * Co + coBase) * K;
    for (int m = 0; m < 16; ++m) {
      const float* wrow = Wb + (size_t)m * K;
      const float* mrow = Mb + (size_t)m * K;
      for (int k = tid; k < K; k += 256) {
        int ci = k & ciMask;
        int r  = k >> ciShift;
        Alds[m * K + k] = f2bf(wrow[ci * 9 + r] * mrow[ci * 9 + r]);
      }
    }
  }
  __syncthreads();

  const int pixBase = (blockIdx.x * 8 + wave) * 64;
  if (pixBase >= HW) return;                 // wave-uniform (latent layer)

  const int hi   = lane >> 4;
  const int nrow = lane & 15;
  const __bf16* xb   = xp + (size_t)b * (H + 2) * Wp * Ci;
  const __bf16* arow = Alds + nrow * K;

  int p[4];
  unsigned eoff[4];                          // per-lane, loop-invariant
#pragma unroll
  for (int j = 0; j < 4; ++j) {
    p[j] = pixBase + nrow + 16 * j;
    int pb = (p[j] >> wdShift) * Wp + (p[j] & (Wd - 1));
    eoff[j] = (unsigned)(pb * Ci + hi * 8);
  }

  v8f acc[4] = {{}, {}, {}, {}};

  for (int k0 = 0; k0 < K; k0 += 32) {
    union alignas(32) { v16bf v; uint4 q[2]; } au;
    au.q[0] = *(const uint4*)(arow + k0 + hi * 8);
    au.q[1] = *(const uint4*)(arow + k0 + hi * 8 + 16);
    v16bf af = au.v;

    int r  = k0 >> ciShift;                  // wave-uniform
    int dy = (r >= 3) + (r >= 6);
    int dx = r - dy * 3;
    unsigned u = (unsigned)((dy * Wp + dx) * Ci + (k0 & ciMask));  // uniform

    union alignas(32) { v16bf v; uint4 q[2]; } bu[4];
#pragma unroll
    for (int j = 0; j < 4; ++j) {
      const __bf16* s = xb + (eoff[j] + u);
      bu[j].q[0] = *(const uint4*)(s);
      bu[j].q[1] = *(const uint4*)(s + 16);
    }
#pragma unroll
    for (int j = 0; j < 4; ++j)
      acc[j] = __builtin_amdgcn_wmma_f32_16x16x32_bf16(
          false, af, false, bu[j].v, (short)0, acc[j], false, false);
  }

  // C/D: VGPR r -> co = coBase + r + 8*hi (contiguous in NHWC) at pixel p[j]
  float4 bq0 = *(const float4*)(bias + coBase + 8 * hi);
  float4 bq1 = *(const float4*)(bias + coBase + 8 * hi + 4);
#pragma unroll
  for (int j = 0; j < 4; ++j) {
    float* d = y + ((size_t)b * HW + p[j]) * Co + coBase + 8 * hi;
    float4 o;
    o.x = acc[j][0] + bq0.x; o.y = acc[j][1] + bq0.y;
    o.z = acc[j][2] + bq0.z; o.w = acc[j][3] + bq0.w;
    *(float4*)(d) = o;
    o.x = acc[j][4] + bq1.x; o.y = acc[j][5] + bq1.y;
    o.z = acc[j][6] + bq1.z; o.w = acc[j][7] + bq1.w;
    *(float4*)(d + 4) = o;
  }
}

// ---------------------------------------------------------------------------
// Instance norm + ReLU over NHWC, in place. grid = (C, B)
// ---------------------------------------------------------------------------
__global__ __launch_bounds__(256) void inorm_relu(float* __restrict__ y, int C, int HW)
{
  int b = blockIdx.y, c = blockIdx.x;
  float* p = y + (size_t)b * HW * C + c;
  float s = 0.f, s2 = 0.f;
  for (int i = threadIdx.x; i < HW; i += 256) { float v = p[(size_t)i * C]; s += v; s2 += v * v; }
  __shared__ float sh0[256], sh1[256];
  sh0[threadIdx.x] = s; sh1[threadIdx.x] = s2;
  __syncthreads();
  for (int o = 128; o > 0; o >>= 1) {
    if (threadIdx.x < o) { sh0[threadIdx.x] += sh0[threadIdx.x + o]; sh1[threadIdx.x] += sh1[threadIdx.x + o]; }
    __syncthreads();
  }
  float inv_n = 1.0f / (float)HW;
  float mu  = sh0[0] * inv_n;
  float var = sh1[0] * inv_n - mu * mu;
  float inv = rsqrtf(var + 1e-5f);
  for (int i = threadIdx.x; i < HW; i += 256) {
    float v = (p[(size_t)i * C] - mu) * inv;
    p[(size_t)i * C] = v > 0.f ? v : 0.f;
  }
}

// ---------------------------------------------------------------------------
__global__ __launch_bounds__(256) void maxpool2k(const float* __restrict__ x,
    float* __restrict__ y, int C, int H, int W, int N)
{
  int i = blockIdx.x * 256 + threadIdx.x; if (i >= N) return;
  int Wo = W >> 1;
  int c  = i % C;  int t = i / C;
  int xo = t % Wo; t /= Wo;
  int yo = t % (H >> 1); int b = t / (H >> 1);
  const float* xb = x + (((size_t)b * H + 2 * yo) * W + 2 * xo) * C + c;
  size_t rs = (size_t)W * C;
  float v = fmaxf(fmaxf(xb[0], xb[C]), fmaxf(xb[rs], xb[rs + C]));
  y[i] = v;
}

// ---------------------------------------------------------------------------
__global__ __launch_bounds__(256) void up2_bilinear(const float* __restrict__ x,
    float* __restrict__ y, int C, int H, int W, int Ctot, int N)
{
  int i = blockIdx.x * 256 + threadIdx.x; if (i >= N) return;
  int Wo = W * 2, Ho = H * 2;
  int c  = i % C;  int t = i / C;
  int xo = t % Wo; t /= Wo;
  int yo = t % Ho; int b = t / Ho;
  float fy = yo * 0.5f - 0.25f;
  float fx = xo * 0.5f - 0.25f;
  int y0 = (int)floorf(fy), x0 = (int)floorf(fx);
  float wy = fy - (float)y0, wx = fx - (float)x0;
  int y0c = imin(imax(y0, 0), H - 1),     y1c = imin(imax(y0 + 1, 0), H - 1);
  int x0c = imin(imax(x0, 0), W - 1),     x1c = imin(imax(x0 + 1, 0), W - 1);
  const float* xb = x + (size_t)b * H * W * C + c;
  float v00 = xb[((size_t)y0c * W + x0c) * C], v01 = xb[((size_t)y0c * W + x1c) * C];
  float v10 = xb[((size_t)y1c * W + x0c) * C], v11 = xb[((size_t)y1c * W + x1c) * C];
  float v = (1.f - wy) * ((1.f - wx) * v00 + wx * v01)
          +        wy  * ((1.f - wx) * v10 + wx * v11);
  y[(((size_t)b * Ho + yo) * Wo + xo) * Ctot + c] = v;
}

__global__ __launch_bounds__(256) void copy_to_concat(const float* __restrict__ s,
    float* __restrict__ y, int Cs, int HW, int Coff, int Ctot, int N)
{
  int i = blockIdx.x * 256 + threadIdx.x; if (i >= N) return;
  int c = i % Cs; int t = i / Cs; int p = t % HW; int b = t / HW;
  y[((size_t)b * HW + p) * Ctot + Coff + c] = s[i];
}

// ---------------------------------------------------------------------------
__global__ __launch_bounds__(256) void head1x1_res(
    const float* __restrict__ x, const float* __restrict__ w1, const float* __restrict__ b1,
    const float* __restrict__ w2, const float* __restrict__ b2,
    const float* __restrict__ w3, const float* __restrict__ b3,
    const float* __restrict__ x0, float* __restrict__ preR, float* __restrict__ preI, int HW)
{
  int p = blockIdx.x * 256 + threadIdx.x; if (p >= HW) return;
  int b = blockIdx.y;
  const float* xb = x + ((size_t)b * HW + p) * 32;
  float o2 = b2[0];
  for (int o = 0; o < 16; ++o) {
    float acc = b1[o];
    for (int c = 0; c < 32; ++c) acc += w1[o * 32 + c] * xb[c];
    o2 += w2[o] * acc;
  }
  float o3 = w3[0] * o2 + b3[0];
  float pred = o3 + x0[(size_t)b * HW + p];
  preR[(size_t)b * HW + p] = pred;
  preI[(size_t)b * HW + p] = 0.f;
}

// ---------------------------------------------------------------------------
// 256-pt radix-2 DIT FFT, rows (is_col=0) / cols (is_col=1).
// ---------------------------------------------------------------------------
__global__ __launch_bounds__(256) void fft256(
    const float* __restrict__ rin, const float* __restrict__ iin,
    float* __restrict__ rout, float* __restrict__ iout, int is_col, float sign)
{
  __shared__ float sre[256], sim[256];
  int t = threadIdx.x;
  int lid = blockIdx.x;
  size_t base; int stride;
  if (is_col) { int b = lid >> 8, c = lid & 255; base = (size_t)b * 65536 + c; stride = 256; }
  else        { base = (size_t)lid * 256; stride = 1; }
  int rt = (int)(__brev((unsigned)t) >> 24);
  sre[rt] = rin[base + (size_t)t * stride];
  sim[rt] = iin[base + (size_t)t * stride];
  __syncthreads();
  for (int len = 2; len <= 256; len <<= 1) {
    int half = len >> 1;
    if (t < 128) {
      int blk = t / half, j = t - blk * half;
      int i0 = blk * len + j, i1 = i0 + half;
      float ang = sign * 6.28318530718f * (float)j / (float)len;
      float sv, cv; __sincosf(ang, &sv, &cv);
      float vr = sre[i1], vi = sim[i1];
      float tr = vr * cv - vi * sv;
      float ti = vr * sv + vi * cv;
      float ur = sre[i0], ui = sim[i0];
      sre[i0] = ur + tr; sim[i0] = ui + ti;
      sre[i1] = ur - tr; sim[i1] = ui - ti;
    }
    __syncthreads();
  }
  rout[base + (size_t)t * stride] = sre[t] * 0.0625f;
  iout[base + (size_t)t * stride] = sim[t] * 0.0625f;
}

__global__ __launch_bounds__(256) void dc_combine(float* __restrict__ kr, float* __restrict__ ki,
    const float* __restrict__ ksp, const float* __restrict__ mask, int N)
{
  int i = blockIdx.x * 256 + threadIdx.x; if (i >= N) return;
  float m  = mask[i];
  float sr = ksp[2 * i], si = ksp[2 * i + 1];
  kr[i] = m * sr + (1.f - m) * kr[i];
  ki[i] = m * si + (1.f - m) * ki[i];
}

__global__ __launch_bounds__(256) void extract_re(const float* __restrict__ re,
    float* __restrict__ out, int N)
{
  int i = blockIdx.x * 256 + threadIdx.x; if (i >= N) return;
  out[i] = re[i];
}

// ---------------------------------------------------------------------------
extern "C" void kernel_launch(void* const* d_in, const int* in_sizes, int n_in,
                              void* d_out, int out_size, void* d_ws, size_t ws_size,
                              hipStream_t stream) {
  (void)in_sizes; (void)n_in; (void)out_size; (void)ws_size;
  const float* x0   = (const float*)d_in[0];
  const float* ksp  = (const float*)d_in[1];
  const float* mask = (const float*)d_in[2];
  const float* dm[4] = {(const float*)d_in[3], (const float*)d_in[6], (const float*)d_in[9],  (const float*)d_in[12]};
  const float* dW[4] = {(const float*)d_in[4], (const float*)d_in[7], (const float*)d_in[10], (const float*)d_in[13]};
  const float* db[4] = {(const float*)d_in[5], (const float*)d_in[8], (const float*)d_in[11], (const float*)d_in[14]};
  const float* lm = (const float*)d_in[15];
  const float* lW = (const float*)d_in[16];
  const float* lb = (const float*)d_in[17];
  const float* um[4] = {(const float*)d_in[18], (const float*)d_in[21], (const float*)d_in[24], (const float*)d_in[27]};
  const float* uW[4] = {(const float*)d_in[19], (const float*)d_in[22], (const float*)d_in[25], (const float*)d_in[28]};
  const float* ub[4] = {(const float*)d_in[20], (const float*)d_in[23], (const float*)d_in[26], (const float*)d_in[29]};
  const float* w1 = (const float*)d_in[30]; const float* b1 = (const float*)d_in[31];
  const float* w2 = (const float*)d_in[32]; const float* b2 = (const float*)d_in[33];
  const float* w3 = (const float*)d_in[34]; const float* b3 = (const float*)d_in[35];
  float* out = (float*)d_out;

  const int B = 8;
  float* F = (float*)d_ws; size_t off = 0;
  auto alloc = [&](size_t n) { float* p = F + off; off += n; return p; };
  float* skip0 = alloc((size_t)B * 32  * 65536);   // NHWC
  float* skip1 = alloc((size_t)B * 64  * 16384);
  float* skip2 = alloc((size_t)B * 128 * 4096);
  float* skip3 = alloc((size_t)B * 256 * 1024);
  float* poolb = alloc((size_t)B * 32  * 16384);
  float* latb  = alloc((size_t)B * 256 * 256);
  float* catb  = alloc((size_t)B * 64  * 65536);
  float* ubuf  = alloc((size_t)B * 32  * 65536);
  float* pAre  = alloc((size_t)B * 65536); float* pAim = alloc((size_t)B * 65536);
  float* pBre  = alloc((size_t)B * 65536); float* pBim = alloc((size_t)B * 65536);
  __bf16* padb = (__bf16*)(F + off); off += ((size_t)B * 64 * 258 * 258 + 1) / 2;

  auto conv = [&](const float* xin, const float* Wt, const float* bias, const float* mod,
                  float* y, int Ci, int Co, int H) {
    int HW = H * H;
    if (Ci == 1) {
      hipLaunchKernelGGL(conv0_direct, dim3(HW / 256, 1, B), dim3(256), 0, stream,
                         xin, Wt, bias, mod, y, H, H);
    } else {
      int Np = B * Ci * (H + 2) * (H + 2);
      hipLaunchKernelGGL(pad_nhwc, dim3((Np + 255) / 256), dim3(256), 0, stream,
                         xin, padb, Ci, H, H, Np);
      int K = Ci * 9;
      size_t shbytes = (size_t)16 * K * sizeof(__bf16);
      int gx = HW / 512; if (gx < 1) gx = 1;
      dim3 g(gx, Co / 16, B);
      hipLaunchKernelGGL(modconv3_wmma, g, dim3(256), shbytes, stream,
                         padb, Wt, bias, mod, y, Ci, __builtin_ctz((unsigned)Ci),
                         Co, H, __builtin_ctz((unsigned)H), K);
    }
    hipLaunchKernelGGL(inorm_relu, dim3(Co, B), dim3(256), 0, stream, y, Co, HW);
  };
  auto pool = [&](const float* xin, float* y, int C, int H) {
    int N = B * C * (H / 2) * (H / 2);
    hipLaunchKernelGGL(maxpool2k, dim3((N + 255) / 256), dim3(256), 0, stream, xin, y, C, H, H, N);
  };
  auto up2 = [&](const float* xin, float* y, int C, int H, int Ctot) {
    int N = B * C * (2 * H) * (2 * H);
    hipLaunchKernelGGL(up2_bilinear, dim3((N + 255) / 256), dim3(256), 0, stream, xin, y, C, H, H, Ctot, N);
  };
  auto cat = [&](const float* s, float* y, int Cs, int HW, int Coff, int Ctot) {
    int N = B * Cs * HW;
    hipLaunchKernelGGL(copy_to_concat, dim3((N + 255) / 256), dim3(256), 0, stream, s, y, Cs, HW, Coff, Ctot, N);
  };

  // ---- encoder ----
  conv(x0,    dW[0], db[0], dm[0], skip0, 1,   32,  256);
  pool(skip0, poolb, 32, 256);
  conv(poolb, dW[1], db[1], dm[1], skip1, 32,  64,  128);
  pool(skip1, poolb, 64, 128);
  conv(poolb, dW[2], db[2], dm[2], skip2, 64,  128, 64);
  pool(skip2, poolb, 128, 64);
  conv(poolb, dW[3], db[3], dm[3], skip3, 128, 256, 32);
  pool(skip3, poolb, 256, 32);
  // ---- latent ----
  conv(poolb, lW, lb, lm, latb, 256, 256, 16);
  // ---- decoder ----
  up2(latb, catb, 256, 16, 512);  cat(skip3, catb, 256, 1024,  256, 512);
  conv(catb, uW[0], ub[0], um[0], ubuf, 512, 128, 32);
  up2(ubuf, catb, 128, 32, 256);  cat(skip2, catb, 128, 4096,  128, 256);
  conv(catb, uW[1], ub[1], um[1], ubuf, 256, 64, 64);
  up2(ubuf, catb, 64, 64, 128);   cat(skip1, catb, 64,  16384, 64,  128);
  conv(catb, uW[2], ub[2], um[2], ubuf, 128, 32, 128);
  up2(ubuf, catb, 32, 128, 64);   cat(skip0, catb, 32,  65536, 32,  64);
  conv(catb, uW[3], ub[3], um[3], ubuf, 64, 32, 256);
  // ---- head + residual -> pred (complex, im=0) ----
  hipLaunchKernelGGL(head1x1_res, dim3(65536 / 256, B), dim3(256), 0, stream,
                     ubuf, w1, b1, w2, b2, w3, b3, x0, pAre, pAim, 65536);
  // ---- fft2 (ortho) -> DC merge -> ifft2 (ortho) ----
  dim3 fg(B * 256), fb(256);
  hipLaunchKernelGGL(fft256, fg, fb, 0, stream, pAre, pAim, pBre, pBim, 0, -1.f);
  hipLaunchKernelGGL(fft256, fg, fb, 0, stream, pBre, pBim, pAre, pAim, 1, -1.f);
  int N = B * 65536;
  hipLaunchKernelGGL(dc_combine, dim3((N + 255) / 256), dim3(256), 0, stream, pAre, pAim, ksp, mask, N);
  hipLaunchKernelGGL(fft256, fg, fb, 0, stream, pAre, pAim, pBre, pBim, 1, 1.f);
  hipLaunchKernelGGL(fft256, fg, fb, 0, stream, pBre, pBim, pAre, pAim, 0, 1.f);
  hipLaunchKernelGGL(extract_re, dim3((N + 255) / 256), dim3(256), 0, stream, pAre, out, N);
}
